// ToroidalAttention_5282809774388
// MI455X (gfx1250) — matrix operand order repaired and checked by reading, not compiled, not run on hardware
//
#include <hip/hip_runtime.h>

typedef __attribute__((ext_vector_type(16))) _Float16 v16h;
typedef __attribute__((ext_vector_type(8)))  _Float16 v8h;
typedef __attribute__((ext_vector_type(8)))  float    v8f;

#define B_  4
#define T_  2048
#define C_  512
#define H_  8
#define DK_ 64

__device__ __forceinline__ int iabs_(int x){ return x < 0 ? -x : x; }
__device__ __forceinline__ int imin_(int a, int b){ return a < b ? a : b; }

// ---------------------------------------------------------------------------
// WMMA fragment loaders (ISA 7.12.2 layouts, wave32)
// A (16x32 f16, row-major, row stride ld): lane<16 -> row=lane, K 0..7 & 16..23
//                                          lane>=16 -> row=lane-16, K 8..15 & 24..31
__device__ __forceinline__ v16h load_a_frag(const _Float16* base, int ld, int row0, int lane) {
  int half = lane >> 4, lr = lane & 15;
  const _Float16* p = base + (size_t)(row0 + lr) * ld + half * 8;
  v8h lo = *(const v8h*)(p);
  v8h hi = *(const v8h*)(p + 16);
  v16h a;
#pragma unroll
  for (int i = 0; i < 8; ++i) { a[i] = lo[i]; a[8 + i] = hi[i]; }
  return a;
}

// B (32x16 f16): lane<16 holds column N=lane, K 0..7 & 16..23; lane>=16 column N=lane-16, K 8..15 & 24..31.
// `bt` points at a matrix whose ROW n is column n of B (contiguous K), row stride ld.
__device__ __forceinline__ v16h load_b_frag(const _Float16* bt, int ld, int col0, int lane) {
  int half = lane >> 4, lr = lane & 15;
  const _Float16* p = bt + (size_t)(col0 + lr) * ld + half * 8;
  v8h lo = *(const v8h*)(p);
  v8h hi = *(const v8h*)(p + 16);
  v16h b;
#pragma unroll
  for (int i = 0; i < 8; ++i) { b[i] = lo[i]; b[8 + i] = hi[i]; }
  return b;
}

__device__ __forceinline__ v8f zero8() { v8f z = {0.f,0.f,0.f,0.f,0.f,0.f,0.f,0.f}; return z; }

// CDNA5 async copy: global -> LDS, 16B per lane, tracked by ASYNCcnt.
// GVS form: mem_addr = SGPR64 + VGPR32(byte off); LDS addr = wave-relative byte off in VGPR.
__device__ __forceinline__ void async_ld_b128(unsigned ldsOff, unsigned gOff, const void* base) {
  asm volatile("global_load_async_to_lds_b128 %0, %1, %2"
               :: "v"(ldsOff), "v"(gOff), "s"((unsigned long long)(uintptr_t)base)
               : "memory");
}
__device__ __forceinline__ void wait_async0() {
  asm volatile("s_wait_asynccnt 0x0" ::: "memory");
}

// ---------------------------------------------------------------------------
// fp32 -> fp16 conversion
__global__ __launch_bounds__(256) void cvt_f32_f16(const float* __restrict__ in,
                                                   _Float16* __restrict__ out, int n) {
  int i = blockIdx.x * 256 + threadIdx.x;
  if (i < n) out[i] = (_Float16)in[i];
}

// ---------------------------------------------------------------------------
// GEMM: Y[R=8192, 512] = A[R,512](f16) @ W^T + bias,  W row n = weights of out col n.
// mode 0: write f16 head-major  [B,H,T,64]
// mode 2: write f16 transposed  [B,H,64,T]
// mode 3: write f32 flat        [B*T, C]
__global__ __launch_bounds__(128) void proj_gemm(const _Float16* __restrict__ A,
                                                 const _Float16* __restrict__ W,
                                                 const float* __restrict__ bias,
                                                 _Float16* __restrict__ outH,
                                                 float* __restrict__ outF,
                                                 int mode) {
  int lane = threadIdx.x & 31;
  int wave = threadIdx.x >> 5;          // 0..3 -> column group
  int half = lane >> 4, lr = lane & 15;
  int t0 = blockIdx.x * 16;             // 16 output rows

  v8f acc[8];
#pragma unroll
  for (int nt = 0; nt < 8; ++nt) acc[nt] = zero8();

  for (int kc = 0; kc < C_; kc += 32) {
    v16h a = load_a_frag(A + kc, C_, t0, lane);
#pragma unroll
    for (int nt = 0; nt < 8; ++nt) {
      int n0 = (wave * 8 + nt) * 16;
      v16h b = load_b_frag(W + kc, C_, n0, lane);
      acc[nt] = __builtin_amdgcn_wmma_f32_16x16x32_f16(false, a, false, b,
                                                       (short)0, acc[nt], false, false);
    }
  }

#pragma unroll
  for (int nt = 0; nt < 8; ++nt) {
    int n0  = (wave * 8 + nt) * 16;
    int col = n0 + lr;
    float bv = bias[col];
#pragma unroll
    for (int r = 0; r < 8; ++r) {
      int row = t0 + r + 8 * half;      // [0, B*T)
      float v = acc[nt][r] + bv;
      if (mode == 3) {
        outF[(size_t)row * C_ + col] = v;
      } else {
        int b = row >> 11, t = row & (T_ - 1);
        int h = col >> 6,  d = col & (DK_ - 1);
        if (mode == 2) outH[(((size_t)(b * H_ + h) * DK_) + d) * T_ + t] = (_Float16)v;
        else           outH[(((size_t)(b * H_ + h) * T_)  + t) * DK_ + d] = (_Float16)v;
      }
    }
  }
}

// ---------------------------------------------------------------------------
// Fused flash attention with multiplicative Tonnetz mask.
// Qh,Kh: [B,H,T,64] f16.  Vt: [B,H,64,T] f16.  Oh out: [B*T, C] f16.
// 4 waves/block share one (b,h); K/V 32-kv chunks double-buffered in LDS via
// global_load_async_to_lds_b128 (load of chunk i+1 overlaps compute of chunk i).
__global__ __launch_bounds__(128) void attn_fused(const _Float16* __restrict__ Qh,
                                                  const _Float16* __restrict__ Kh,
                                                  const _Float16* __restrict__ Vt,
                                                  _Float16* __restrict__ Oh) {
  __shared__ __attribute__((aligned(16))) _Float16 Ks[2][32 * 64]; // [j_local][d]
  __shared__ __attribute__((aligned(16))) _Float16 Vs[2][64 * 32]; // [d][j_local]
  __shared__ __attribute__((aligned(16))) _Float16 pt[4][16 * 48]; // per-wave P tile, pitch 48
  __shared__ float mtab[16];                                       // Tonnetz mask table, d<=12

  int tid  = threadIdx.x;
  int lane = tid & 31;
  int wave = tid >> 5;
  int half = lane >> 4, lr = lane & 15;

  if (tid < 16) {
    int d = tid;
    mtab[d] = (d <= 2) ? 1.0f : __expf(-(float)d);
  }

  int bh = blockIdx.y;                          // b*H + h
  int q0 = (blockIdx.x * 4 + wave) * 16;        // 16 query rows per wave

  const _Float16* Qb = Qh + (size_t)bh * T_ * DK_;
  const _Float16* Kb = Kh + (size_t)bh * T_ * DK_;
  const _Float16* Vb = Vt + (size_t)bh * DK_ * T_;

  // ---- stage chunk j0 into LDS buffer buf (4KB K + 4KB V, 4 x b128 per thread)
  auto stage = [&](int j0, int buf) {
    unsigned kbase = (unsigned)(uintptr_t)(&Ks[buf][0]);
    unsigned vbase = (unsigned)(uintptr_t)(&Vs[buf][0]);
    const void* kg = (const void*)(Kb + (size_t)j0 * DK_);  // 32 rows x 128B, contiguous
    const void* vg = (const void*)(Vb + j0);                // 64 rows x 64B, row stride T_*2
#pragma unroll
    for (int i = 0; i < 2; ++i) {
      unsigned off = (unsigned)(tid + i * 128) * 16u;
      async_ld_b128(kbase + off, off, kg);
    }
#pragma unroll
    for (int i = 0; i < 2; ++i) {
      int idx = tid + i * 128;                  // 0..255
      int d = idx >> 2, part = idx & 3;
      async_ld_b128(vbase + (unsigned)idx * 16u,
                    (unsigned)(d * T_ * 2 + part * 16), vg);
    }
  };

  v16h aq0 = load_a_frag(Qb + 0,  DK_, q0, lane);  // d 0..31
  v16h aq1 = load_a_frag(Qb + 32, DK_, q0, lane);  // d 32..63

  int xq[8], yq[8];
#pragma unroll
  for (int r = 0; r < 8; ++r) {
    int qi = q0 + r + 8 * half;
    xq[r] = qi % 12;
    yq[r] = (qi / 12) % 12;
  }

  float m[8], lsum[8];
  v8f acc[4];
#pragma unroll
  for (int r = 0; r < 8; ++r) { m[r] = -1e30f; lsum[r] = 0.0f; }
#pragma unroll
  for (int dt = 0; dt < 4; ++dt) acc[dt] = zero8();

  stage(0, 0);                                   // prologue: prefetch chunk 0

  for (int j0 = 0; j0 < T_; j0 += 32) {
    int buf = (j0 >> 5) & 1;
    wait_async0();                               // chunk j0 landed (this wave's copies)
    __syncthreads();                             // all waves' copies landed, all done with buf^1
    if (j0 + 32 < T_) stage(j0 + 32, buf ^ 1);   // overlap next chunk's loads with compute
    const _Float16* KsB = &Ks[buf][0];
    const _Float16* VsB = &Vs[buf][0];

    // ---- scores: two 16x16 j-subtiles, K = d_k = 64 (2 chunks of 32), from LDS
    v8f s0 = zero8(), s1 = zero8();
    s0 = __builtin_amdgcn_wmma_f32_16x16x32_f16(false, aq0, false,
          load_b_frag(KsB + 0,  DK_, 0, lane),  (short)0, s0, false, false);
    s0 = __builtin_amdgcn_wmma_f32_16x16x32_f16(false, aq1, false,
          load_b_frag(KsB + 32, DK_, 0, lane),  (short)0, s0, false, false);
    s1 = __builtin_amdgcn_wmma_f32_16x16x32_f16(false, aq0, false,
          load_b_frag(KsB + 0,  DK_, 16, lane), (short)0, s1, false, false);
    s1 = __builtin_amdgcn_wmma_f32_16x16x32_f16(false, aq1, false,
          load_b_frag(KsB + 32, DK_, 16, lane), (short)0, s1, false, false);

    // ---- multiplicative Tonnetz mask + 1/sqrt(dk), then online softmax
    int c0 = j0 + lr, c1 = j0 + 16 + lr;
    int x0 = c0 % 12, y0 = (c0 / 12) % 12;
    int x1 = c1 % 12, y1 = (c1 / 12) % 12;

    float mn[8];
#pragma unroll
    for (int r = 0; r < 8; ++r) {
      int dx0 = iabs_(xq[r] - x0); dx0 = imin_(dx0, 12 - dx0);
      int dy0 = iabs_(yq[r] - y0); dy0 = imin_(dy0, 12 - dy0);
      int dx1 = iabs_(xq[r] - x1); dx1 = imin_(dx1, 12 - dx1);
      int dy1 = iabs_(yq[r] - y1); dy1 = imin_(dy1, 12 - dy1);
      s0[r] = s0[r] * 0.125f * mtab[dx0 + dy0];
      s1[r] = s1[r] * 0.125f * mtab[dx1 + dy1];
      float t = fmaxf(s0[r], s1[r]);                 // row max over this 32-chunk
      t = fmaxf(t, __shfl_xor(t, 1, 32));
      t = fmaxf(t, __shfl_xor(t, 2, 32));
      t = fmaxf(t, __shfl_xor(t, 4, 32));
      t = fmaxf(t, __shfl_xor(t, 8, 32));
      mn[r] = fmaxf(m[r], t);
    }

#pragma unroll
    for (int r = 0; r < 8; ++r) {
      float sc = __expf(m[r] - mn[r]);
      float p0 = __expf(s0[r] - mn[r]);
      float p1 = __expf(s1[r] - mn[r]);
      float ps = p0 + p1;
      ps += __shfl_xor(ps, 1, 32);
      ps += __shfl_xor(ps, 2, 32);
      ps += __shfl_xor(ps, 4, 32);
      ps += __shfl_xor(ps, 8, 32);
      lsum[r] = lsum[r] * sc + ps;
      m[r] = mn[r];
#pragma unroll
      for (int dt = 0; dt < 4; ++dt) acc[dt][r] *= sc;
      int prow = (r + 8 * half) * 48;
      pt[wave][prow + lr]      = (_Float16)p0;
      pt[wave][prow + 16 + lr] = (_Float16)p1;
    }
    asm volatile("s_wait_dscnt 0" ::: "memory");     // LDS store -> load turn (same wave)

    // ---- reload P as an A fragment (16x32)
    v16h pa;
    {
      const _Float16* pp = &pt[wave][lr * 48 + half * 8];
      v8h lo = *(const v8h*)(pp);
      v8h hi = *(const v8h*)(pp + 16);
#pragma unroll
      for (int i = 0; i < 8; ++i) { pa[i] = lo[i]; pa[8 + i] = hi[i]; }
    }

    // ---- O += P @ V  (dk = 64 -> four 16-col N tiles), B from transposed V in LDS
#pragma unroll
    for (int dt = 0; dt < 4; ++dt) {
      v16h bv = load_b_frag(VsB, 32, dt * 16, lane);
      acc[dt] = __builtin_amdgcn_wmma_f32_16x16x32_f16(false, pa, false, bv,
                                                       (short)0, acc[dt], false, false);
    }
  }

  // ---- epilogue: normalize, write f16 [B*T, C] (head h occupies cols h*64..)
  int b = bh >> 3, h = bh & 7;
#pragma unroll
  for (int dt = 0; dt < 4; ++dt) {
#pragma unroll
    for (int r = 0; r < 8; ++r) {
      float v = acc[dt][r] / lsum[r];
      int t = q0 + r + 8 * half;
      Oh[((size_t)(b * T_ + t)) * C_ + h * DK_ + dt * 16 + lr] = (_Float16)v;
    }
  }
}

// ---------------------------------------------------------------------------
extern "C" void kernel_launch(void* const* d_in, const int* in_sizes, int n_in,
                              void* d_out, int out_size, void* d_ws, size_t ws_size,
                              hipStream_t stream) {
  (void)in_sizes; (void)n_in; (void)out_size; (void)ws_size;
  const float* x  = (const float*)d_in[0];
  const float* Wq = (const float*)d_in[1];
  const float* bq = (const float*)d_in[2];
  const float* Wk = (const float*)d_in[3];
  const float* bk = (const float*)d_in[4];
  const float* Wv = (const float*)d_in[5];
  const float* bv = (const float*)d_in[6];
  const float* Wo = (const float*)d_in[7];
  const float* bo = (const float*)d_in[8];

  char* ws = (char*)d_ws;
  const size_t NX = (size_t)B_ * T_ * C_;     // 4,194,304
  const size_t NW = (size_t)C_ * C_;          // 262,144
  const size_t NH = (size_t)B_ * H_ * T_ * DK_;

  _Float16* xh  = (_Float16*)(ws);
  _Float16* wqh = (_Float16*)(ws + 2 * NX);
  _Float16* wkh = wqh + NW;
  _Float16* wvh = wkh + NW;
  _Float16* woh = wvh + NW;
  _Float16* Qh  = woh + NW;
  _Float16* Kh  = Qh + NH;
  _Float16* Vt  = Kh + NH;
  _Float16* Oh  = Vt + NH;

  cvt_f32_f16<<<(int)((NX + 255) / 256), 256, 0, stream>>>(x,  xh,  (int)NX);
  cvt_f32_f16<<<(int)((NW + 255) / 256), 256, 0, stream>>>(Wq, wqh, (int)NW);
  cvt_f32_f16<<<(int)((NW + 255) / 256), 256, 0, stream>>>(Wk, wkh, (int)NW);
  cvt_f32_f16<<<(int)((NW + 255) / 256), 256, 0, stream>>>(Wv, wvh, (int)NW);
  cvt_f32_f16<<<(int)((NW + 255) / 256), 256, 0, stream>>>(Wo, woh, (int)NW);

  dim3 gproj(B_ * T_ / 16);                   // 512 row tiles, 4 waves cover 512 cols
  proj_gemm<<<gproj, 128, 0, stream>>>(xh, wqh, bq, Qh, nullptr, 0);
  proj_gemm<<<gproj, 128, 0, stream>>>(xh, wkh, bk, Kh, nullptr, 0);
  proj_gemm<<<gproj, 128, 0, stream>>>(xh, wvh, bv, Vt, nullptr, 2);

  dim3 gatt(T_ / 64, B_ * H_);                // 4 q-tiles per block (1 per wave)
  attn_fused<<<gatt, 128, 0, stream>>>(Qh, Kh, Vt, Oh);

  proj_gemm<<<gproj, 128, 0, stream>>>(Oh, woh, bo, nullptr, (float*)d_out, 3);
}
